// ChunkedLocalAttention_70128226009221
// MI455X (gfx1250) — matrix operand order repaired and checked by reading, not compiled, not run on hardware
//
#include <hip/hip_runtime.h>
#include <hip/hip_bf16.h>

// ---------------------------------------------------------------------------
// Chunked local attention w/ landmarks on gfx1250 (MI455X), bf16 WMMA path.
// B=4, S=8192, D=768, H=12, HD=64, CS=512, NL=32, n_chunks=16, ctx=544.
// ---------------------------------------------------------------------------

typedef __bf16 bf16_t;
typedef __attribute__((ext_vector_type(16))) __bf16 v16bf;
typedef __attribute__((ext_vector_type(8)))  __bf16 v8bf;
typedef __attribute__((ext_vector_type(8)))  float  v8f;

#define DD   768
#define HH   12
#define HD   64
#define BB   4
#define SS   8192
#define CS_  512
#define NL_  32
#define NCH  16

#define LDSP 40   // padded LDS row pitch (bf16 elems) for P staging

static __device__ __forceinline__ v16bf cat8(v8bf lo, v8bf hi) {
  return __builtin_shufflevector(lo, hi, 0,1,2,3,4,5,6,7,8,9,10,11,12,13,14,15);
}

static __device__ __forceinline__ v8f wmma_bf16(v16bf a, v16bf b, v8f c) {
  return __builtin_amdgcn_wmma_f32_16x16x32_bf16(false, a, false, b, (short)0, c,
                                                 false, false);
}

// ---------------------------------------------------------------------------
// f32 -> bf16 elementwise convert
// ---------------------------------------------------------------------------
__global__ void cvt_f32_bf16(const float* __restrict__ in,
                             bf16_t* __restrict__ out, int n) {
  int i = blockIdx.x * 256 + threadIdx.x;
  if (i < n) out[i] = (bf16_t)in[i];
}

// ---------------------------------------------------------------------------
// landmarks: mean-pool 256-token segments of x -> (B, NL, D) bf16
// flat segment t = b*NL + i starts at token row t*256.
// ---------------------------------------------------------------------------
__global__ void landmark_kernel(const float* __restrict__ x,
                                bf16_t* __restrict__ lm) {
  int idx = blockIdx.x * 256 + threadIdx.x;       // over B*NL*D = 98304
  if (idx >= BB * NL_ * DD) return;
  int d = idx % DD;
  int t = idx / DD;
  const float* p = x + (size_t)t * 256 * DD + d;
  float s = 0.f;
  #pragma unroll 8
  for (int j = 0; j < 256; ++j) s += p[(size_t)j * DD];
  lm[idx] = (bf16_t)(s * (1.0f / 256.0f));
}

// ---------------------------------------------------------------------------
// bf16 WMMA GEMM: out(M x 768) = A(M x 768) @ W^T + bias
//   W is row-major (N=768, K=768) -> B fragment = contiguous 32B of W.
//   128 threads = 4 waves; wave computes a 64x64 tile (4x4 WMMA tiles).
// mode 0: out bf16, layout (B, H, rpb, HD)      [Q / K]  (rpb = 1<<rpbShift)
// mode 1: out bf16, layout (B, H, HD, rpb)      [V transposed, packed stores]
// mode 2: out f32 , row-major M x 768           [final projection]
// ---------------------------------------------------------------------------
__global__ __launch_bounds__(128)
void gemm_wmma(const bf16_t* __restrict__ A, const bf16_t* __restrict__ W,
               const float* __restrict__ bias, void* __restrict__ outp,
               int mode, int rpbShift) {
  const int lane  = threadIdx.x & 31;
  const int wave  = threadIdx.x >> 5;      // 0..3
  const int col16 = lane & 15;
  const int hi    = (lane >> 4) & 1;
  const int blockM = blockIdx.y * 128 + (wave >> 1) * 64;
  const int blockN = blockIdx.x * 128 + (wave & 1) * 64;

  v8f c[4][4] = {};

  for (int k0 = 0; k0 < DD; k0 += 32) {
    v16bf a[4], b[4];
    #pragma unroll
    for (int mt = 0; mt < 4; ++mt) {
      int m = blockM + mt * 16 + col16;
      const bf16_t* p = A + (size_t)m * DD + k0 + hi * 16;
      a[mt] = cat8(*(const v8bf*)p, *(const v8bf*)(p + 8));
    }
    #pragma unroll
    for (int nt = 0; nt < 4; ++nt) {
      int n = blockN + nt * 16 + col16;
      b[nt] = *(const v16bf*)(W + (size_t)n * DD + k0 + hi * 16);
    }
    #pragma unroll
    for (int mt = 0; mt < 4; ++mt)
      #pragma unroll
      for (int nt = 0; nt < 4; ++nt)
        c[mt][nt] = wmma_bf16(a[mt], b[nt], c[mt][nt]);
  }

  // epilogue: C layout is row = r + hi*8 (per VGPR r), col = col16
  const int rpbMask = (1 << rpbShift) - 1;
  #pragma unroll
  for (int nt = 0; nt < 4; ++nt) {
    int   n  = blockN + nt * 16 + col16;
    float bv = bias[n];
    int   h  = n >> 6;
    int   hd = n & 63;
    #pragma unroll
    for (int mt = 0; mt < 4; ++mt) {
      int m0 = blockM + mt * 16 + hi * 8;   // first of this lane's 8 rows
      if (mode == 2) {
        float* dst = (float*)outp + (size_t)m0 * DD + n;
        #pragma unroll
        for (int r = 0; r < 8; ++r)
          dst[(size_t)r * DD] = c[mt][nt][r] + bv;
      } else {
        int bb = m0 >> rpbShift;            // 8-row groups never straddle bb
        int s0 = m0 & rpbMask;
        if (mode == 0) {
          bf16_t* dst = (bf16_t*)outp +
              ((((size_t)(bb * HH + h) << rpbShift) + s0) * HD + hd);
          #pragma unroll
          for (int r = 0; r < 8; ++r)
            dst[(size_t)r * HD] = (bf16_t)(c[mt][nt][r] + bv);
        } else {
          v8bf pk;
          #pragma unroll
          for (int r = 0; r < 8; ++r) pk[r] = (bf16_t)(c[mt][nt][r] + bv);
          bf16_t* dst = (bf16_t*)outp +
              ((((size_t)(bb * HH + h) * HD + hd) << rpbShift) + s0);
          *(v8bf*)dst = pk;                 // 16B packed store (s0 % 8 == 0)
        }
      }
    }
  }
}

// ---------------------------------------------------------------------------
// Attention: per (b, chunk, head, qblock). 256 threads = 8 waves.
// Each wave owns 16 q rows, streams ctx = 32 landmarks + 512 chunk keys
// in 17 steps of 32 keys with online softmax (flash style).
//   Q, K : (B, H, S, HD) bf16 row-major
//   Vt   : (B, H, HD, S) bf16 (transposed)
//   Klm  : (B, H, NL, HD), Vlm : (B, H, HD, NL)
//   Aout : (B, S, D) bf16
// ---------------------------------------------------------------------------
__global__ __launch_bounds__(256)
void attn_wmma(const bf16_t* __restrict__ Q,  const bf16_t* __restrict__ K,
               const bf16_t* __restrict__ Vt, const bf16_t* __restrict__ Klm,
               const bf16_t* __restrict__ Vlm, bf16_t* __restrict__ Aout) {
  int idx = blockIdx.x;
  const int qb = idx & 3;      idx >>= 2;
  const int h  = idx % HH;     idx /= HH;
  const int ch = idx % NCH;    idx /= NCH;
  const int b  = idx;

  const int lane  = threadIdx.x & 31;
  const int wave  = threadIdx.x >> 5;   // 0..7
  const int col16 = lane & 15;
  const int hi    = (lane >> 4) & 1;
  const int bh    = b * HH + h;
  const int srow0 = ch * CS_ + qb * 128 + wave * 16;

  __shared__ bf16_t ldsP[8][16 * LDSP];  // per-wave padded 16x32 P staging
  bf16_t* myP = &ldsP[wave][0];

  // Q A-fragments (16 rows x 64 hd = 2 k-steps of 32)
  v16bf qa[2];
  {
    const bf16_t* qbase = Q + ((size_t)bh * SS + srow0 + col16) * HD;
    #pragma unroll
    for (int ks = 0; ks < 2; ++ks) {
      const bf16_t* p = qbase + ks * 32 + hi * 16;
      qa[ks] = cat8(*(const v8bf*)p, *(const v8bf*)(p + 8));
    }
  }

  float rm[8], rs[8], alpha[8];
  #pragma unroll
  for (int r = 0; r < 8; ++r) { rm[r] = -1e30f; rs[r] = 0.f; }
  v8f o[4] = {};

  const float scale = 0.125f;   // HD^-0.5

  for (int step = 0; step < 17; ++step) {
    const bf16_t* ksrc;
    const bf16_t* vsrc;
    int vpitch;
    if (step == 0) {
      ksrc   = Klm + (size_t)bh * NL_ * HD;
      vsrc   = Vlm + (size_t)bh * HD * NL_;
      vpitch = NL_;
    } else {
      int kb = ch * CS_ + (step - 1) * 32;
      ksrc   = K  + ((size_t)bh * SS + kb) * HD;
      vsrc   = Vt + ((size_t)bh * HD) * SS + kb;
      vpitch = SS;
    }

    // scores: S(16x32) = Q(16x64) @ K^T ; two 16-col subtiles, 2 k-steps each
    v8f s0 = {}, s1 = {};
    #pragma unroll
    for (int ks = 0; ks < 2; ++ks) {
      v16bf b0 = *(const v16bf*)(ksrc + (size_t)(col16)      * HD + ks * 32 + hi * 16);
      v16bf b1 = *(const v16bf*)(ksrc + (size_t)(16 + col16) * HD + ks * 32 + hi * 16);
      s0 = wmma_bf16(qa[ks], b0, s0);
      s1 = wmma_bf16(qa[ks], b1, s1);
    }

    // online softmax per row (row = r + hi*8; each row lives in one 16-lane half)
    #pragma unroll
    for (int r = 0; r < 8; ++r) {
      float v0 = s0[r] * scale;
      float v1 = s1[r] * scale;
      float mx = fmaxf(v0, v1);
      #pragma unroll
      for (int msk = 1; msk < 16; msk <<= 1)
        mx = fmaxf(mx, __shfl_xor(mx, msk, 32));
      float mnew = fmaxf(rm[r], mx);
      float a    = __expf(rm[r] - mnew);
      float p0   = __expf(v0 - mnew);
      float p1   = __expf(v1 - mnew);
      float sm   = p0 + p1;
      #pragma unroll
      for (int msk = 1; msk < 16; msk <<= 1)
        sm += __shfl_xor(sm, msk, 32);
      rs[r]    = rs[r] * a + sm;
      rm[r]    = mnew;
      alpha[r] = a;
      int row = hi * 8 + r;
      myP[row * LDSP + col16]      = (bf16_t)p0;  // C-layout -> LDS row-major
      myP[row * LDSP + 16 + col16] = (bf16_t)p1;
    }

    // rescale running output by alpha (per row = per C VGPR)
    #pragma unroll
    for (int t = 0; t < 4; ++t)
      #pragma unroll
      for (int r = 0; r < 8; ++r)
        o[t][r] *= alpha[r];

    // P as A-fragment (16x32) from LDS (padded pitch, loads stay 16B-aligned)
    v16bf pa;
    {
      const bf16_t* p = myP + col16 * LDSP + hi * 16;
      pa = cat8(*(const v8bf*)p, *(const v8bf*)(p + 8));
    }

    // O(16x64) += P(16x32) @ V(32x64): 4 B-fragments from transposed V
    #pragma unroll
    for (int t = 0; t < 4; ++t) {
      int hd = t * 16 + col16;
      v16bf vb = *(const v16bf*)(vsrc + (size_t)hd * vpitch + hi * 16);
      o[t] = wmma_bf16(pa, vb, o[t]);
    }
  }

  // normalize and scatter to (B, S, D) bf16
  #pragma unroll
  for (int t = 0; t < 4; ++t) {
    #pragma unroll
    for (int r = 0; r < 8; ++r) {
      int   row = srow0 + hi * 8 + r;
      float val = o[t][r] / rs[r];
      Aout[((size_t)b * SS + row) * DD + h * HD + t * 16 + col16] = (bf16_t)val;
    }
  }
}

// ---------------------------------------------------------------------------
// host launcher
// ---------------------------------------------------------------------------
static inline bf16_t* ws_alloc(char* base, size_t& off, size_t nelem) {
  bf16_t* p = (bf16_t*)(base + off);
  off += (nelem * sizeof(bf16_t) + 255) & ~(size_t)255;
  return p;
}

extern "C" void kernel_launch(void* const* d_in, const int* in_sizes, int n_in,
                              void* d_out, int out_size, void* d_ws, size_t ws_size,
                              hipStream_t stream) {
  (void)in_sizes; (void)n_in; (void)out_size; (void)ws_size;

  const float* x   = (const float*)d_in[0];
  const float* q_w = (const float*)d_in[1];
  const float* q_b = (const float*)d_in[2];
  const float* k_w = (const float*)d_in[3];
  const float* k_b = (const float*)d_in[4];
  const float* v_w = (const float*)d_in[5];
  const float* v_b = (const float*)d_in[6];
  const float* o_w = (const float*)d_in[7];
  const float* o_b = (const float*)d_in[8];

  const size_t NX  = (size_t)BB * SS * DD;         // 25165824
  const size_t NW  = (size_t)DD * DD;              // 589824
  const size_t NLM = (size_t)BB * NL_ * DD;        // 98304
  const size_t NQ  = (size_t)BB * HH * SS * HD;    // == NX
  const size_t NKL = (size_t)BB * HH * NL_ * HD;   // 98304

  char* ws = (char*)d_ws;
  size_t off = 0;
  bf16_t* xb   = ws_alloc(ws, off, NX);
  bf16_t* qwb  = ws_alloc(ws, off, NW);
  bf16_t* kwb  = ws_alloc(ws, off, NW);
  bf16_t* vwb  = ws_alloc(ws, off, NW);
  bf16_t* owb  = ws_alloc(ws, off, NW);
  bf16_t* lmb  = ws_alloc(ws, off, NLM);
  bf16_t* Qb   = ws_alloc(ws, off, NQ);
  bf16_t* Kb   = ws_alloc(ws, off, NQ);
  bf16_t* Vtb  = ws_alloc(ws, off, NQ);
  bf16_t* Klm  = ws_alloc(ws, off, NKL);
  bf16_t* Vlm  = ws_alloc(ws, off, NKL);
  bf16_t* AOut = ws_alloc(ws, off, NX);

  // 1) converts
  cvt_f32_bf16<<<(int)((NX + 255) / 256), 256, 0, stream>>>(x,   xb,  (int)NX);
  cvt_f32_bf16<<<(int)((NW + 255) / 256), 256, 0, stream>>>(q_w, qwb, (int)NW);
  cvt_f32_bf16<<<(int)((NW + 255) / 256), 256, 0, stream>>>(k_w, kwb, (int)NW);
  cvt_f32_bf16<<<(int)((NW + 255) / 256), 256, 0, stream>>>(v_w, vwb, (int)NW);
  cvt_f32_bf16<<<(int)((NW + 255) / 256), 256, 0, stream>>>(o_w, owb, (int)NW);

  // 2) landmarks
  landmark_kernel<<<(int)((NLM + 255) / 256), 256, 0, stream>>>(x, lmb);

  // 3) projections over x: M = B*S = 32768, N = 768 (rpbShift = 13: S = 8192)
  {
    dim3 grid(DD / 128, (BB * SS) / 128);
    gemm_wmma<<<grid, 128, 0, stream>>>(xb, qwb, q_b, Qb,  0, 13);
    gemm_wmma<<<grid, 128, 0, stream>>>(xb, kwb, k_b, Kb,  0, 13);
    gemm_wmma<<<grid, 128, 0, stream>>>(xb, vwb, v_b, Vtb, 1, 13);
  }
  // 4) projections over landmarks: M = B*NL = 128 (rpbShift = 5: NL = 32)
  {
    dim3 grid(DD / 128, (BB * NL_) / 128);
    gemm_wmma<<<grid, 128, 0, stream>>>(lmb, kwb, k_b, Klm, 0, 5);
    gemm_wmma<<<grid, 128, 0, stream>>>(lmb, vwb, v_b, Vlm, 1, 5);
  }

  // 5) attention: grid = B * NCH * H * 4 qblocks = 3072 blocks of 256 threads
  attn_wmma<<<BB * NCH * HH * 4, 256, 0, stream>>>(Qb, Kb, Vtb, Klm, Vlm, AOut);

  // 6) output projection -> f32 d_out
  {
    dim3 grid(DD / 128, (BB * SS) / 128);
    gemm_wmma<<<grid, 128, 0, stream>>>(AOut, owb, o_b, d_out, 2, 13);
  }
}
